// decoder_59528246722726
// MI455X (gfx1250) — matrix operand order repaired and checked by reading, not compiled, not run on hardware
//
#include <hip/hip_runtime.h>
#include <cstdint>

// ---------------------------------------------------------------------------
// Problem constants (match reference)
// ---------------------------------------------------------------------------
#define VOCAB   32000
#define EMB     512
#define HID     512
#define H3      1536
#define BATCH   16
#define SLEN    64
#define TSTEPS  128
#define START_ID 1

typedef __bf16 bf16;
typedef unsigned short ushort_t;
typedef __attribute__((ext_vector_type(8)))  float v8f;
typedef __attribute__((ext_vector_type(16))) bf16  v16bf;
typedef __attribute__((ext_vector_type(8)))  bf16  v8bf;
typedef __attribute__((ext_vector_type(4)))  unsigned int u32x4_t;
typedef __attribute__((ext_vector_type(8)))  int i32x8_t;
typedef __attribute__((ext_vector_type(4)))  int i32x4_t;

// ---------------------------------------------------------------------------
// WMMA helpers (CDNA5 gfx1250: V_WMMA_F32_16X16X32_BF16, wave32)
// ---------------------------------------------------------------------------
__device__ inline v8f zero8() {
  v8f z;
#pragma unroll
  for (int i = 0; i < 8; ++i) z[i] = 0.0f;
  return z;
}

__device__ inline v8f wmma_bf16(v16bf a, v16bf b, v8f c) {
  // 8 args: (neg_a, A, neg_b, B, c_mod, C, reuse_a, reuse_b)
  return __builtin_amdgcn_wmma_f32_16x16x32_bf16(false, a, false, b,
                                                 (short)0, c, false, false);
}

__device__ inline v16bf join16(v8bf lo, v8bf hi) {
  v16bf r;
#pragma unroll
  for (int i = 0; i < 8; ++i) { r[i] = lo[i]; r[i + 8] = hi[i]; }
  return r;
}

// A-matrix fragment (16x32 bf16, MxK). Row-major source A[M][lda].
// ISA layout: lane<16 holds row=lane, K = {k0..k0+7} U {k0+16..k0+23};
//             lane>=16 holds row=lane-16, K = {k0+8..k0+15} U {k0+24..k0+31}.
__device__ inline v16bf load_a(const bf16* A, int lda, int m_base, int k0, int lane) {
  const bf16* p = A + (size_t)(m_base + (lane & 15)) * lda + k0 + ((lane & 16) ? 8 : 0);
  v8bf lo = *reinterpret_cast<const v8bf*>(p);
  v8bf hi = *reinterpret_cast<const v8bf*>(p + 16);
  return join16(lo, hi);
}

// B-matrix fragment (32x16 bf16, KxN) for X @ W.T: column n of B = row n of W.
// W row-major [N][ldk]. lane<16: col=lane, K={k0..k0+7} U {k0+16..k0+23};
// lane>=16: col=lane-16, K={k0+8..k0+15} U {k0+24..k0+31} -> two b128 loads.
__device__ inline v16bf load_b(const bf16* W, int ldk, int n_base, int k0, int lane) {
  const bf16* p = W + (size_t)(n_base + (lane & 15)) * ldk + k0 + ((lane & 16) ? 8 : 0);
  v8bf lo = *reinterpret_cast<const v8bf*>(p);
  v8bf hi = *reinterpret_cast<const v8bf*>(p + 16);
  return join16(lo, hi);
}

__device__ inline float sigmoidf_(float x) { return 1.0f / (1.0f + __expf(-x)); }

// ---------------------------------------------------------------------------
// Tensor Data Mover: async global->LDS tile DMA (ISA ch.8 D# layout).
// Loads a 2D tile of 128 rows x 32 bf16 (64B/row) from a row-major tensor
// whose rows are 512 bf16 long.  LDS destination gets 16B padding every 64B
// (pad_interval=16 DWORDs, pad_amount=4 DWORDs) -> 80B row pitch, which makes
// the 16-row-strided ds_load_b128 fragment reads bank-conflict-free.
// ---------------------------------------------------------------------------
#define LTILE_PITCH  40                        // bf16 elems per padded LDS row
#define LTILE_BYTES  (128 * LTILE_PITCH * 2)   // 10240 B per tile

__device__ inline unsigned lds_off(const void* p) {
  // generic pointer into LDS: addr[31:0] is the DS byte offset (aperture map)
  return (unsigned)(uintptr_t)p;
}

__device__ inline void tdm_load_tile_128x32(unsigned lds_byte, const bf16* gsrc,
                                            unsigned tensor_rows) {
  unsigned long long ga = (unsigned long long)(uintptr_t)gsrc;
  u32x4_t g0;
  g0[0] = 1u;                                            // count=1, user D#
  g0[1] = lds_byte;                                      // lds_addr (bytes)
  g0[2] = (unsigned)ga;                                  // global_addr[31:0]
  g0[3] = (unsigned)((ga >> 32) & 0x1FFFFFFull)          // global_addr[56:32]
          | (2u << 30);                                  // type = 2 ("image")
  i32x8_t g1;
  g1[0] = (1 << 16)                                      // data_size = 2 bytes
        | (1 << 20)                                      // pad_enable
        | (3 << 22)                                      // pad_interval: 16 DW
        | (3 << 25);                                     // pad_amount:   4 DW
  g1[1] = (int)((512u & 0xFFFFu) << 16);                 // tensor_dim0 = 512
  g1[2] = (int)(((tensor_rows & 0xFFFFu) << 16));        // tensor_dim1 lo
  g1[3] = (int)(((tensor_rows >> 16) & 0xFFFFu)          // tensor_dim1 hi
        | (32u << 16));                                  // tile_dim0 = 32
  g1[4] = 128;                                           // tile_dim1 = 128
  g1[5] = 512;                                           // tensor_dim0_stride
  g1[6] = 0;
  g1[7] = 0;                                             // dim1 stride unused (2D)
  i32x4_t gz;
  gz[0] = 0; gz[1] = 0; gz[2] = 0; gz[3] = 0;            // groups 2/3: 2D tile
  i32x8_t gz8;
  gz8[0] = 0; gz8[1] = 0; gz8[2] = 0; gz8[3] = 0;
  gz8[4] = 0; gz8[5] = 0; gz8[6] = 0; gz8[7] = 0;        // extra group (6-arg ABI)
  // this toolchain's builtin is the 6-arg form:
  // (uint32x4 g0, int32x8 g1, int32x4, int32x4, int32x8, i32 cpol)
  __builtin_amdgcn_tensor_load_to_lds(g0, g1, gz, gz, gz8, 0);
}

// ---------------------------------------------------------------------------
// Kernel: fp32 -> bf16 weight conversion
// ---------------------------------------------------------------------------
__global__ void cvt_bf16_kernel(const float* __restrict__ src,
                                ushort_t* __restrict__ dst, int n) {
  int i = blockIdx.x * blockDim.x + threadIdx.x;
  if (i < n) reinterpret_cast<bf16*>(dst)[i] = (bf16)src[i];
}

// ---------------------------------------------------------------------------
// Kernel: gather teacher-forcing embeddings for all steps, convert to bf16.
// ---------------------------------------------------------------------------
__global__ void embed_gather_kernel(const float* __restrict__ embed,
                                    const int* __restrict__ dec_outs,
                                    ushort_t* __restrict__ emb_out) {
  int row = blockIdx.x;            // 0 .. T*B-1
  int t = row >> 4;
  int b = row & 15;
  int tok = (t == 0) ? START_ID : dec_outs[b * TSTEPS + (t - 1)];
  const float* s = embed + (size_t)tok * EMB;
  bf16* d = reinterpret_cast<bf16*>(emb_out) + (size_t)row * EMB;
  for (int j = threadIdx.x; j < EMB; j += blockDim.x) d[j] = (bf16)s[j];
}

// ---------------------------------------------------------------------------
// Persistent recurrence kernel: one workgroup (32 wave32s) runs all 128 steps.
// LDS layout (bytes):
//   h      fp32 [16][512]   @      0   ( 32768)
//   gsum   fp32 [16][1024]  @  32768   ( 65536)   gi+gh for r,z gates
//   gin    fp32 [16][512]   @  98304   ( 32768)   i_n
//   ghn    fp32 [16][512]   @ 131072   ( 32768)   h_n
//   hb16   bf16 [16][512]   @ 163840   ( 16384)
//   comb   bf16 [16][1536]  @ 180224   ( 49152)   [att1|att2|op]
//   sc1    fp32 [16][64]    @ 229376   (  4096)
//   sc2    fp32 [16][64]    @ 233472   (  4096)
// ---------------------------------------------------------------------------
#define RECUR_LDS_BYTES 237568

__global__ __launch_bounds__(1024)
void recur_kernel(const ushort_t* __restrict__ emb_b,
                  const ushort_t* __restrict__ wih_b,
                  const ushort_t* __restrict__ whh_b,
                  const ushort_t* __restrict__ hw_b,
                  const float* __restrict__ h1, const float* __restrict__ h2,
                  const float* __restrict__ encW, const float* __restrict__ encb,
                  const float* __restrict__ bih, const float* __restrict__ bhh,
                  const float* __restrict__ hb,
                  const float* __restrict__ l1, const float* __restrict__ l2,
                  const unsigned char* __restrict__ mask1,
                  const unsigned char* __restrict__ mask2,
                  ushort_t* __restrict__ hall_b) {
  extern __shared__ char smem[];
  float* sh_h    = reinterpret_cast<float*>(smem);
  float* sh_gsum = reinterpret_cast<float*>(smem + 32768);
  float* sh_gin  = reinterpret_cast<float*>(smem + 98304);
  float* sh_ghn  = reinterpret_cast<float*>(smem + 131072);
  bf16*  sh_hb   = reinterpret_cast<bf16*>(smem + 163840);
  bf16*  sh_comb = reinterpret_cast<bf16*>(smem + 180224);
  float* sh_sc1  = reinterpret_cast<float*>(smem + 229376);
  float* sh_sc2  = reinterpret_cast<float*>(smem + 233472);

  const int tid  = threadIdx.x;
  const int lane = tid & 31;
  const int wave = tid >> 5;

  // ---- h0 = tanh(cat(h1,h2) @ encW.T + encb)  (one-time, fp32 VALU)
  for (int idx = tid; idx < BATCH * HID; idx += 1024) {
    int b = idx >> 9, j = idx & 511;
    const float* w  = encW + (size_t)j * (2 * HID);
    const float* x1 = h1 + b * HID;
    const float* x2 = h2 + b * HID;
    float acc = encb[j];
    for (int k = 0; k < HID; ++k) acc += x1[k] * w[k];
    for (int k = 0; k < HID; ++k) acc += x2[k] * w[HID + k];
    sh_h[idx] = tanhf(acc);
  }
  __syncthreads();

  for (int t = 0; t < TSTEPS; ++t) {
    // -- P0: bf16 shadow of h (WMMA A operand for Whh GEMM)
    for (int idx = tid; idx < BATCH * HID; idx += 1024)
      sh_hb[idx] = (bf16)sh_h[idx];
    __syncthreads();

    // -- P1: gate GEMMs  gi = emb@Wih.T ; gh = h@Whh.T
    //    K-outer: one wave owns N-tiles {wave, wave+32, wave+64}; shared A
    //    fragments are loaded once per K-step and feed 6 WMMAs.
    {
      const bf16* embA = reinterpret_cast<const bf16*>(emb_b) +
                         (size_t)t * BATCH * EMB;
      const bf16* Wih = reinterpret_cast<const bf16*>(wih_b);
      const bf16* Whh = reinterpret_cast<const bf16*>(whh_b);
      const int n0 = wave * 16;            // tile 0..31   -> gsum cols 0..511
      const int n1 = (wave + 32) * 16;     // tile 32..63  -> gsum cols 512..1023
      const int n2 = (wave + 64) * 16;     // tile 64..95  -> gin/ghn
      v8f ai0 = zero8(), ai1 = zero8(), ai2 = zero8();
      v8f ah0 = zero8(), ah1 = zero8(), ah2 = zero8();
      for (int kk = 0; kk < 16; ++kk) {
        const int k0 = kk * 32;
        v16bf ae  = load_a(embA, EMB, 0, k0, lane);
        v16bf ahf = load_a(sh_hb, HID, 0, k0, lane);
        v16bf w0 = load_b(Wih, EMB, n0, k0, lane);
        v16bf w1 = load_b(Wih, EMB, n1, k0, lane);
        v16bf w2 = load_b(Wih, EMB, n2, k0, lane);
        v16bf u0 = load_b(Whh, HID, n0, k0, lane);
        v16bf u1 = load_b(Whh, HID, n1, k0, lane);
        v16bf u2 = load_b(Whh, HID, n2, k0, lane);
        ai0 = wmma_bf16(ae, w0, ai0);
        ai1 = wmma_bf16(ae, w1, ai1);
        ai2 = wmma_bf16(ae, w2, ai2);
        ah0 = wmma_bf16(ahf, u0, ah0);
        ah1 = wmma_bf16(ahf, u1, ah1);
        ah2 = wmma_bf16(ahf, u2, ah2);
      }
      const int cl = lane & 15;
      const int rbase = (lane & 16) ? 8 : 0;
      const int col0 = n0 + cl;
      const int col1 = n1 + cl;
      const int c2 = n2 + cl - 1024;
#pragma unroll
      for (int r = 0; r < 8; ++r) {
        sh_gsum[(rbase + r) * 1024 + col0] = ai0[r] + ah0[r];
        sh_gsum[(rbase + r) * 1024 + col1] = ai1[r] + ah1[r];
        sh_gin[(rbase + r) * 512 + c2] = ai2[r];
        sh_ghn[(rbase + r) * 512 + c2] = ah2[r];
      }
    }
    __syncthreads();

    // -- P2: GRU gates (-> comb[:,1024:]) + attention scores (vs OLD h)
    for (int idx = tid; idx < BATCH * HID; idx += 1024) {
      int b = idx >> 9, j = idx & 511;
      float r = sigmoidf_(sh_gsum[b * 1024 + j] + bih[j] + bhh[j]);
      float z = sigmoidf_(sh_gsum[b * 1024 + 512 + j] + bih[512 + j] + bhh[512 + j]);
      float n = tanhf(sh_gin[b * 512 + j] + bih[1024 + j] +
                      r * (sh_ghn[b * 512 + j] + bhh[1024 + j]));
      float op = (1.0f - z) * n + z * sh_h[b * 512 + j];
      sh_comb[b * H3 + 1024 + j] = (bf16)op;
    }
    {
      int b = tid >> 6, s = tid & 63;     // 1024 threads = 16*64 scores
      const float* hrow = sh_h + b * HID;
      const float* p1 = l1 + ((size_t)b * SLEN + s) * HID;
      const float* p2 = l2 + ((size_t)b * SLEN + s) * HID;
      float s1 = 0.0f, s2 = 0.0f;
      for (int k = 0; k < HID; ++k) { s1 += p1[k] * hrow[k]; s2 += p2[k] * hrow[k]; }
      sh_sc1[tid] = mask1[b * SLEN + s] ? -1e9f : s1;
      sh_sc2[tid] = mask2[b * SLEN + s] ? -1e9f : s2;
    }
    __syncthreads();

    // -- P3a: softmax over S=64 per batch row
    if (tid < BATCH) {
#pragma unroll
      for (int which = 0; which < 2; ++which) {
        float* sc = which ? sh_sc2 : sh_sc1;
        float m = -1e30f;
        for (int s = 0; s < SLEN; ++s) m = fmaxf(m, sc[tid * SLEN + s]);
        float sum = 0.0f;
        for (int s = 0; s < SLEN; ++s) {
          float e = __expf(sc[tid * SLEN + s] - m);
          sc[tid * SLEN + s] = e;
          sum += e;
        }
        float inv = 1.0f / sum;
        for (int s = 0; s < SLEN; ++s) sc[tid * SLEN + s] *= inv;
      }
    }
    __syncthreads();

    // -- P3b: contexts -> comb[:,0:512] and comb[:,512:1024]
    for (int idx = tid; idx < BATCH * HID; idx += 1024) {
      int b = idx >> 9, j = idx & 511;
      const float* a1 = sh_sc1 + b * SLEN;
      const float* a2 = sh_sc2 + b * SLEN;
      const float* p1 = l1 + (size_t)b * SLEN * HID + j;
      const float* p2 = l2 + (size_t)b * SLEN * HID + j;
      float c1 = 0.0f, c2 = 0.0f;
      for (int s = 0; s < SLEN; ++s) {
        c1 += a1[s] * p1[(size_t)s * HID];
        c2 += a2[s] * p2[(size_t)s * HID];
      }
      sh_comb[b * H3 + j]       = (bf16)c1;
      sh_comb[b * H3 + 512 + j] = (bf16)c2;
    }
    __syncthreads();

    // -- P4: h_new = tanh(comb @ hW.T + hb)  (one N-tile per wave, K=1536)
    {
      const bf16* Whw = reinterpret_cast<const bf16*>(hw_b);
      const int n0 = wave * 16;
      v8f acc = zero8();
      v16bf a  = load_a(sh_comb, H3, 0, 0, lane);
      v16bf bw = load_b(Whw, H3, n0, 0, lane);
      for (int kk = 0; kk < 47; ++kk) {
        const int kn = (kk + 1) * 32;
        v16bf na = load_a(sh_comb, H3, 0, kn, lane);
        v16bf nb = load_b(Whw, H3, n0, kn, lane);
        acc = wmma_bf16(a, bw, acc);
        a = na; bw = nb;
      }
      acc = wmma_bf16(a, bw, acc);

      const int col = n0 + (lane & 15);
      const int rbase = (lane & 16) ? 8 : 0;
      bf16* hall = reinterpret_cast<bf16*>(hall_b);
#pragma unroll
      for (int r = 0; r < 8; ++r) {
        int m = rbase + r;
        float v = tanhf(acc[r] + hb[col]);
        sh_h[m * HID + col] = v;                                  // next-step h
        hall[((size_t)(t * BATCH + m)) * HID + col] = (bf16)v;    // for logits
      }
    }
    __syncthreads();
  }
}

// ---------------------------------------------------------------------------
// Logits GEMM: out[T*B, V] = Hall[2048,512](bf16) @ gW[V,512](bf16).T + gb
// 256 threads (8 waves). Workgroup tile 128M x 128N; per wave 32M x 64N.
// TDM double-buffered pipeline: while the 8 waves run this K-step's WMMAs
// from LDS, wave 0's TENSOR_LOAD_TO_LDS DMAs stream the next 128x32 A and B
// slabs into the alternate LDS buffers (TENSORcnt-tracked). gW bf16 (32.8 MB)
// is L2-resident on MI455X (192 MB L2), so the DMAs are L2 reads.
// ---------------------------------------------------------------------------
__global__ __launch_bounds__(256)
void logits_kernel(const ushort_t* __restrict__ hall_b,
                   const ushort_t* __restrict__ gw_b,
                   const float* __restrict__ gb,
                   float* __restrict__ out) {
  __shared__ char lsm[4 * LTILE_BYTES];   // A0 | A1 | B0 | B1 (padded tiles)
  bf16* Atile[2] = { reinterpret_cast<bf16*>(lsm),
                     reinterpret_cast<bf16*>(lsm + LTILE_BYTES) };
  bf16* Btile[2] = { reinterpret_cast<bf16*>(lsm + 2 * LTILE_BYTES),
                     reinterpret_cast<bf16*>(lsm + 3 * LTILE_BYTES) };

  const int lane = threadIdx.x & 31;
  const int wave = threadIdx.x >> 5;
  const int wm = wave & 3;        // 4 M-waves (32 rows each)
  const int wn = wave >> 2;       // 2 N-waves (64 cols each)
  const int Mblk = blockIdx.y * 128;
  const int Nblk = blockIdx.x * 128;
  const bf16* A = reinterpret_cast<const bf16*>(hall_b);
  const bf16* W = reinterpret_cast<const bf16*>(gw_b);

  v8f acc[2][4];
#pragma unroll
  for (int mf = 0; mf < 2; ++mf)
#pragma unroll
    for (int nf = 0; nf < 4; ++nf) acc[mf][nf] = zero8();

  // prologue: DMA K-step 0 tiles, drain TENSORcnt, publish via barrier
  if (wave == 0) {
    tdm_load_tile_128x32(lds_off(Atile[0]), A + (size_t)Mblk * HID, TSTEPS * BATCH);
    tdm_load_tile_128x32(lds_off(Btile[0]), W + (size_t)Nblk * HID, VOCAB);
    __builtin_amdgcn_s_wait_tensorcnt(0);
  }
  __syncthreads();

  for (int kk = 0; kk < 16; ++kk) {
    const int cur = kk & 1;
    // kick off next slab's DMAs under this slab's compute
    if (kk < 15 && wave == 0) {
      const int kn = (kk + 1) * 32;
      tdm_load_tile_128x32(lds_off(Atile[cur ^ 1]),
                           A + (size_t)Mblk * HID + kn, TSTEPS * BATCH);
      tdm_load_tile_128x32(lds_off(Btile[cur ^ 1]),
                           W + (size_t)Nblk * HID + kn, VOCAB);
    }

    v16bf a0 = load_a(Atile[cur], LTILE_PITCH, wm * 32,      0, lane);
    v16bf a1 = load_a(Atile[cur], LTILE_PITCH, wm * 32 + 16, 0, lane);
#pragma unroll
    for (int nf = 0; nf < 4; ++nf) {
      v16bf bwf = load_b(Btile[cur], LTILE_PITCH, wn * 64 + nf * 16, 0, lane);
      acc[0][nf] = wmma_bf16(a0, bwf, acc[0][nf]);
      acc[1][nf] = wmma_bf16(a1, bwf, acc[1][nf]);
    }

    // next buffers ready (wave0 drains TENSORcnt), everyone done reading cur
    if (wave == 0) __builtin_amdgcn_s_wait_tensorcnt(0);
    __syncthreads();
  }

  const int rbase = (lane & 16) ? 8 : 0;
  const int cl = lane & 15;
#pragma unroll
  for (int mf = 0; mf < 2; ++mf) {
#pragma unroll
    for (int nf = 0; nf < 4; ++nf) {
      int col = Nblk + wn * 64 + nf * 16 + cl;
      float gbv = gb[col];
#pragma unroll
      for (int r = 0; r < 8; ++r) {
        int row = Mblk + wm * 32 + mf * 16 + rbase + r;
        out[(size_t)row * VOCAB + col] = acc[mf][nf][r] + gbv;
      }
    }
  }
}

// ---------------------------------------------------------------------------
// Host launch
// ---------------------------------------------------------------------------
extern "C" void kernel_launch(void* const* d_in, const int* in_sizes, int n_in,
                              void* d_out, int out_size, void* d_ws, size_t ws_size,
                              hipStream_t stream) {
  (void)in_sizes; (void)n_in; (void)out_size; (void)ws_size;

  const float* h1    = (const float*)d_in[0];
  const float* h2    = (const float*)d_in[1];
  const float* l1    = (const float*)d_in[2];
  const float* l2    = (const float*)d_in[3];
  const float* embed = (const float*)d_in[4];
  const float* Wih   = (const float*)d_in[5];
  const float* Whh   = (const float*)d_in[6];
  const float* bih   = (const float*)d_in[7];
  const float* bhh   = (const float*)d_in[8];
  const float* encW  = (const float*)d_in[9];
  const float* encb  = (const float*)d_in[10];
  const float* hW    = (const float*)d_in[11];
  const float* hb    = (const float*)d_in[12];
  const float* gW    = (const float*)d_in[13];
  const float* gb    = (const float*)d_in[14];
  const int*   dec   = (const int*)d_in[15];
  const unsigned char* mask1 = (const unsigned char*)d_in[16];
  const unsigned char* mask2 = (const unsigned char*)d_in[17];
  float* out = (float*)d_out;

  // Workspace layout (bytes)
  char* ws = (char*)d_ws;
  ushort_t* emb_b  = (ushort_t*)(ws + 0);          // [T*B][E]  bf16  2 MB
  ushort_t* wih_b  = (ushort_t*)(ws + 2097152);    // [3H][E]   bf16  1.5 MB
  ushort_t* whh_b  = (ushort_t*)(ws + 3670016);    // [3H][H]   bf16  1.5 MB
  ushort_t* hw_b   = (ushort_t*)(ws + 5242880);    // [H][3H]   bf16  1.5 MB
  ushort_t* gw_b   = (ushort_t*)(ws + 6815744);    // [V][H]    bf16  32.8 MB
  ushort_t* hall_b = (ushort_t*)(ws + 39583744);   // [T*B][H]  bf16  2 MB

  // 1) Weight fp32 -> bf16 conversions
  {
    int n1 = H3 * EMB;      // Wih
    int n2 = H3 * HID;      // Whh
    int n3 = HID * H3;      // hW
    int n4 = VOCAB * HID;   // gW
    cvt_bf16_kernel<<<(n1 + 255) / 256, 256, 0, stream>>>(Wih, wih_b, n1);
    cvt_bf16_kernel<<<(n2 + 255) / 256, 256, 0, stream>>>(Whh, whh_b, n2);
    cvt_bf16_kernel<<<(n3 + 255) / 256, 256, 0, stream>>>(hW, hw_b, n3);
    cvt_bf16_kernel<<<(n4 + 255) / 256, 256, 0, stream>>>(gW, gw_b, n4);
  }

  // 2) Teacher-forcing embedding gather (+bf16)
  embed_gather_kernel<<<TSTEPS * BATCH, 128, 0, stream>>>(embed, dec, emb_b);

  // 3) Serial recurrence: one persistent workgroup, big dynamic LDS
  (void)hipFuncSetAttribute((const void*)recur_kernel,
                            hipFuncAttributeMaxDynamicSharedMemorySize,
                            RECUR_LDS_BYTES);
  recur_kernel<<<1, 1024, RECUR_LDS_BYTES, stream>>>(
      emb_b, wih_b, whh_b, hw_b, h1, h2, encW, encb, bih, bhh, hb,
      l1, l2, mask1, mask2, hall_b);

  // 4) Big parallel logits GEMM: [2048 x 32000], K=512, TDM-fed WMMA
  dim3 grid(VOCAB / 128, (TSTEPS * BATCH) / 128);   // 250 x 16
  logits_kernel<<<grid, 256, 0, stream>>>(hall_b, gw_b, gb, out);
}
